// RAPStrategy_73667279061356
// MI455X (gfx1250) — compile-verified, matrix operand order checked by reading, not run on hardware
//
#include <hip/hip_runtime.h>
#include <hip/hip_bf16.h>

// ---------------------------------------------------------------------------
// Problem constants (match reference)
// ---------------------------------------------------------------------------
#define B_TOK   65536
#define HID     1024
#define HALF_   512
#define OUT_    1024
#define CTX     64
#define NEXP    4
#define KTOT    1088            // HID + CTX, concatenated K for GEMM1

// Tiling
#define BM      64              // tokens per workgroup
#define KC1     32              // GEMM1 K chunk
#define NKCH1   (KTOT / KC1)    // 34 chunks
#define KC2     32              // GEMM2 K chunk
#define NKCH2   (HALF_ / KC2)   // 16 chunks
#define NT1     16              // GEMM1 N-tiles per wave
#define NT2     32              // GEMM2 N-tiles per wave
#define A_STRIDE  1096          // 1088 + 8 pad (bf16 elems)
#define H_STRIDE  520           // 512 + 8 pad
#define W_STRIDE  40            // 32 + 8 pad (both GEMM weight slabs)

// LDS layout (bytes)
#define A_OFF   0
#define A_BYTES (BM * A_STRIDE * 2)                 // 140288
#define H_OFF   (A_OFF + A_BYTES)
#define H_BYTES (BM * H_STRIDE * 2)                 // 66560
#define W_OFF   (H_OFF + H_BYTES)                   // 206848
#define W_BYTES (1024 * W_STRIDE * 2)               // 81920: [1024 rows][KC+pad]
#define CW_OFF  (W_OFF + W_BYTES)                   // 288768
#define SMEM_BYTES (CW_OFF + BM * 4 * 4)            // 289792 (< 320KB WGP LDS)

typedef __attribute__((ext_vector_type(16))) __bf16 v16bf;
typedef __attribute__((ext_vector_type(8)))  __bf16 v8bf;
typedef __attribute__((ext_vector_type(4)))  __bf16 v4bf;
typedef __attribute__((ext_vector_type(8)))  float  v8f;

// ---------------------------------------------------------------------------
// WMMA helpers (CDNA5: V_WMMA_F32_16X16X32_BF16, wave32)
// ---------------------------------------------------------------------------
__device__ __forceinline__ v8f wmma_bf16(v16bf a, v16bf b, v8f c) {
    return __builtin_amdgcn_wmma_f32_16x16x32_bf16(
        /*neg_a=*/false, a, /*neg_b=*/false, b,
        /*c_mod=*/(short)0, c, /*reuse_a=*/false, /*reuse_b=*/false);
}

// 16-bit A/B fragment per ISA 7.12.2: lane holds two contiguous 8-elem runs:
// K in [kBase + lhalf*8, +8) and [kBase + 16 + lhalf*8, +8).
__device__ __forceinline__ v16bf load_frag(const __bf16* base, int row,
                                           int rowStride, int kBase, int lhalf) {
    const __bf16* p = base + row * rowStride + kBase + lhalf * 8;
    v8bf lo = *(const v8bf*)(p);
    v8bf hi = *(const v8bf*)(p + 16);
    return __builtin_shufflevector(lo, hi, 0,1,2,3,4,5,6,7,8,9,10,11,12,13,14,15);
}

// ---------------------------------------------------------------------------
// Kernel 1: weight precision conversion fp32 -> bf16 into workspace
//   w1[e][n][0:1088] = {super_Wh[e][n][0:1024] , super_Wc[e][n][0:64]}
//   w2[e][o][0:512]  = final_W[e][o][0:512]
// Weight rows are K-major == WMMA B-matrix friendly.
// ---------------------------------------------------------------------------
__global__ void prep_weights(const float* __restrict__ Wh,
                             const float* __restrict__ Wc,
                             const float* __restrict__ Fw,
                             __bf16* __restrict__ w1,
                             __bf16* __restrict__ w2) {
    const long N1 = (long)NEXP * HALF_ * KTOT;
    const long N2 = (long)NEXP * OUT_ * HALF_;
    for (long i = (long)blockIdx.x * blockDim.x + threadIdx.x; i < N1 + N2;
         i += (long)gridDim.x * blockDim.x) {
        if (i < N1) {
            long e = i / ((long)HALF_ * KTOT);
            long rem = i - e * (long)HALF_ * KTOT;
            long n = rem / KTOT, k = rem - n * KTOT;
            float v = (k < HID) ? Wh[(e * HALF_ + n) * HID + k]
                                : Wc[(e * HALF_ + n) * CTX + (k - HID)];
            w1[i] = (__bf16)v;
        } else {
            long j = i - N1;
            w2[j] = (__bf16)Fw[j];
        }
    }
}

// ---------------------------------------------------------------------------
// Kernel 2: exact fp32 gate. One wave per token: 4 dot products (K=1024),
// butterfly reduce, softmax over 4, top-2 renormalized combine weights.
// ---------------------------------------------------------------------------
__global__ void gate_kernel(const float* __restrict__ hidden,
                            const float* __restrict__ gateW,
                            const float* __restrict__ gateB,
                            float* __restrict__ probsOut,
                            float* __restrict__ cwOut) {
    const int wid = threadIdx.x >> 5, lid = threadIdx.x & 31;
    const long b = (long)blockIdx.x * 8 + wid;
    const float* h = hidden + b * HID;

    float a0 = 0.f, a1 = 0.f, a2 = 0.f, a3 = 0.f;
    for (int k = lid; k < HID; k += 32) {
        float hv = h[k];
        a0 += hv * gateW[k];
        a1 += hv * gateW[HID + k];
        a2 += hv * gateW[2 * HID + k];
        a3 += hv * gateW[3 * HID + k];
    }
    for (int off = 16; off; off >>= 1) {
        a0 += __shfl_xor(a0, off, 32);
        a1 += __shfl_xor(a1, off, 32);
        a2 += __shfl_xor(a2, off, 32);
        a3 += __shfl_xor(a3, off, 32);
    }
    if (lid == 0) {
        float l[4] = {a0 + gateB[0], a1 + gateB[1], a2 + gateB[2], a3 + gateB[3]};
        float mx = fmaxf(fmaxf(l[0], l[1]), fmaxf(l[2], l[3]));
        float p[4], s = 0.f;
        #pragma unroll
        for (int e = 0; e < 4; ++e) { p[e] = __expf(l[e] - mx); s += p[e]; }
        float inv = 1.f / s;
        #pragma unroll
        for (int e = 0; e < 4; ++e) { p[e] *= inv; probsOut[b * 4 + e] = p[e]; }
        int i1 = 0;
        #pragma unroll
        for (int e = 1; e < 4; ++e) if (p[e] > p[i1]) i1 = e;
        int i2 = -1;
        #pragma unroll
        for (int e = 0; e < 4; ++e)
            if (e != i1 && (i2 < 0 || p[e] > p[i2])) i2 = e;
        float s2 = p[i1] + p[i2] + 1e-8f;
        float w[4] = {0.f, 0.f, 0.f, 0.f};
        w[i1] = p[i1] / s2;
        w[i2] = p[i2] / s2;
        #pragma unroll
        for (int e = 0; e < 4; ++e) cwOut[b * 4 + e] = w[e];
    }
}

// ---------------------------------------------------------------------------
// Kernel 3: fused expert MLP + combine.
// Both GEMMs stream over K with a full-width weight slab in LDS, so every
// K-step shares ONE A-fragment across all of the wave's N-tiles
// (~1.03 ds_load pairs per WMMA). Next chunk is prefetched into registers
// while the current chunk computes. B-fragments use a 3-deep rotation to
// stay outside the WMMA->VALU WAR hazard window.
// ---------------------------------------------------------------------------
__global__ __launch_bounds__(256)
void moe_main(const float* __restrict__ hidden,
              const float* __restrict__ context,
              const __bf16* __restrict__ w1,
              const __bf16* __restrict__ w2,
              const float* __restrict__ super_b,
              const float* __restrict__ final_b,
              const float* __restrict__ cw,
              float* __restrict__ out) {
    extern __shared__ __align__(16) char smem[];
    __bf16* aT  = (__bf16*)(smem + A_OFF);   // [64][1096] tokens x (hid||ctx)
    __bf16* hT  = (__bf16*)(smem + H_OFF);   // [64][520]  relu'd, w-scaled h
    __bf16* wT  = (__bf16*)(smem + W_OFF);   // [rows][40] weight K-chunk slab
    float*  cwL = (float*)(smem + CW_OFF);   // [64][4] combine weights

    const int tid   = threadIdx.x;
    const int lid   = tid & 31, wid = tid >> 5;
    const int lhalf = lid >> 4, l16 = lid & 15;
    const int token0 = blockIdx.x * BM;
    const int mBase  = (wid & 3) * 16;       // wave's 16-token row block
    const int wgrp   = wid >> 2;             // 0..1: N-tile interleave group

    // ---- stage A-tile: hidden (f32 -> bf16), context, combine weights ----
    for (int i = tid; i < BM * HID / 4; i += 256) {
        int r = i >> 8, kq = (i & 255) * 4;
        float4 v = *(const float4*)(hidden + (size_t)(token0 + r) * HID + kq);
        v4bf b; b.x = (__bf16)v.x; b.y = (__bf16)v.y; b.z = (__bf16)v.z; b.w = (__bf16)v.w;
        *(v4bf*)(aT + r * A_STRIDE + kq) = b;
    }
    for (int i = tid; i < BM * CTX / 4; i += 256) {
        int r = i >> 4, cq = (i & 15) * 4;
        float4 v = *(const float4*)(context + (size_t)(token0 + r) * CTX + cq);
        v4bf b; b.x = (__bf16)v.x; b.y = (__bf16)v.y; b.z = (__bf16)v.z; b.w = (__bf16)v.w;
        *(v4bf*)(aT + r * A_STRIDE + HID + cq) = b;
    }
    for (int i = tid; i < BM * 4; i += 256) cwL[i] = cw[(size_t)token0 * 4 + i];

    // Persistent output accumulators: 32 N-tiles/wave = 256 VGPRs/lane
    v8f acc2[NT2];
    #pragma unroll
    for (int nt = 0; nt < NT2; ++nt)
        acc2[nt] = v8f{0.f, 0.f, 0.f, 0.f, 0.f, 0.f, 0.f, 0.f};

    for (int e = 0; e < NEXP; ++e) {
        // ========== GEMM1: h[64,512] = A[64,1088] x W1^T, K-streamed ==========
        v8f acc1[NT1];
        #pragma unroll
        for (int nt = 0; nt < NT1; ++nt)
            acc1[nt] = v8f{0.f, 0.f, 0.f, 0.f, 0.f, 0.f, 0.f, 0.f};

        const __bf16* w1e = w1 + (size_t)e * HALF_ * KTOT;
        __syncthreads();                  // prior wT/hT users finished
        // prologue: stage chunk 0 (512 rows x KC1)
        #pragma unroll
        for (int j = 0; j < 8; ++j) {
            int id = tid + 256 * j;
            int n = id >> 2, c8 = (id & 3) * 8;
            v8bf v = *(const v8bf*)(w1e + (size_t)n * KTOT + c8);
            *(v8bf*)(wT + n * W_STRIDE + c8) = v;
        }
        __syncthreads();
        for (int kc = 0; kc < NKCH1; ++kc) {
            // issue next chunk's global loads early (hidden behind WMMAs)
            v8bf stg[8];
            const bool more = (kc + 1 < NKCH1);
            if (more) {
                #pragma unroll
                for (int j = 0; j < 8; ++j) {
                    int id = tid + 256 * j;
                    int n = id >> 2, c8 = (id & 3) * 8;
                    stg[j] = *(const v8bf*)(w1e + (size_t)n * KTOT + (kc + 1) * KC1 + c8);
                }
            }
            // compute chunk kc: shared A-fragment, 3-deep B rotation
            v16bf af = load_frag(aT, mBase + l16, A_STRIDE, kc * KC1, lhalf);
            v16bf b0 = load_frag(wT, (0 * 2 + wgrp) * 16 + l16, W_STRIDE, 0, lhalf);
            v16bf b1 = load_frag(wT, (1 * 2 + wgrp) * 16 + l16, W_STRIDE, 0, lhalf);
            #pragma unroll
            for (int nt = 0; nt < NT1; ++nt) {
                v16bf b2 = b0;
                if (nt + 2 < NT1)
                    b2 = load_frag(wT, ((nt + 2) * 2 + wgrp) * 16 + l16,
                                   W_STRIDE, 0, lhalf);
                acc1[nt] = wmma_bf16(af, b0, acc1[nt]);
                b0 = b1; b1 = b2;
            }
            __syncthreads();              // all waves done reading chunk kc
            if (more) {
                #pragma unroll
                for (int j = 0; j < 8; ++j) {
                    int id = tid + 256 * j;
                    int n = id >> 2, c8 = (id & 3) * 8;
                    *(v8bf*)(wT + n * W_STRIDE + c8) = stg[j];
                }
            }
            __syncthreads();              // chunk kc+1 visible
        }
        // epilogue: bias + ReLU + combine-weight scaling -> bf16 hT
        #pragma unroll
        for (int nt = 0; nt < NT1; ++nt) {
            int n = (nt * 2 + wgrp) * 16 + l16;
            float sb = super_b[e * HALF_ + n];
            #pragma unroll
            for (int r = 0; r < 8; ++r) {
                int m = mBase + lhalf * 8 + r;          // C/D: M = r + 8*half
                float x = acc1[nt][r] + sb;
                x = x > 0.f ? x : 0.f;
                x *= cwL[m * 4 + e];
                hT[m * H_STRIDE + n] = (__bf16)x;
            }
        }

        // ========== GEMM2: out[64,1024] += h[64,512] x W2^T, K-streamed ==========
        const __bf16* w2e = w2 + (size_t)e * OUT_ * HALF_;
        __syncthreads();                  // hT writes visible; wT readers done
        // prologue: stage chunk 0 (1024 rows x KC2)
        #pragma unroll
        for (int j = 0; j < 16; ++j) {
            int id = tid + 256 * j;
            int o = id >> 2, c8 = (id & 3) * 8;
            v8bf v = *(const v8bf*)(w2e + (size_t)o * HALF_ + c8);
            *(v8bf*)(wT + o * W_STRIDE + c8) = v;
        }
        __syncthreads();
        for (int kc = 0; kc < NKCH2; ++kc) {
            v8bf stg[16];
            const bool more = (kc + 1 < NKCH2);
            if (more) {
                #pragma unroll
                for (int j = 0; j < 16; ++j) {
                    int id = tid + 256 * j;
                    int o = id >> 2, c8 = (id & 3) * 8;
                    stg[j] = *(const v8bf*)(w2e + (size_t)o * HALF_ + (kc + 1) * KC2 + c8);
                }
            }
            // compute chunk kc: ONE shared A-fragment for 32 WMMAs
            v16bf af = load_frag(hT, mBase + l16, H_STRIDE, kc * KC2, lhalf);
            v16bf b0 = load_frag(wT, (0 * 2 + wgrp) * 16 + l16, W_STRIDE, 0, lhalf);
            v16bf b1 = load_frag(wT, (1 * 2 + wgrp) * 16 + l16, W_STRIDE, 0, lhalf);
            #pragma unroll
            for (int nt = 0; nt < NT2; ++nt) {
                v16bf b2 = b0;
                if (nt + 2 < NT2)
                    b2 = load_frag(wT, ((nt + 2) * 2 + wgrp) * 16 + l16,
                                   W_STRIDE, 0, lhalf);
                acc2[nt] = wmma_bf16(af, b0, acc2[nt]);
                b0 = b1; b1 = b2;
            }
            __syncthreads();
            if (more) {
                #pragma unroll
                for (int j = 0; j < 16; ++j) {
                    int id = tid + 256 * j;
                    int o = id >> 2, c8 = (id & 3) * 8;
                    *(v8bf*)(wT + o * W_STRIDE + c8) = stg[j];
                }
            }
            __syncthreads();
        }
    }

    // ---- epilogue: add combine-weighted final_b, write fp32 output ----
    #pragma unroll
    for (int nt = 0; nt < NT2; ++nt) {
        int o = (nt * 2 + wgrp) * 16 + l16;
        float fb0 = final_b[0 * OUT_ + o], fb1 = final_b[1 * OUT_ + o];
        float fb2 = final_b[2 * OUT_ + o], fb3 = final_b[3 * OUT_ + o];
        #pragma unroll
        for (int r = 0; r < 8; ++r) {
            int m = mBase + lhalf * 8 + r;
            float bias = cwL[m * 4 + 0] * fb0 + cwL[m * 4 + 1] * fb1 +
                         cwL[m * 4 + 2] * fb2 + cwL[m * 4 + 3] * fb3;
            out[(size_t)(token0 + m) * OUT_ + o] = acc2[nt][r] + bias;
        }
    }
}

// ---------------------------------------------------------------------------
extern "C" void kernel_launch(void* const* d_in, const int* in_sizes, int n_in,
                              void* d_out, int out_size, void* d_ws, size_t ws_size,
                              hipStream_t stream) {
    const float* hidden   = (const float*)d_in[0];
    const float* context  = (const float*)d_in[1];
    const float* gate_W   = (const float*)d_in[2];
    const float* gate_b   = (const float*)d_in[3];
    const float* super_Wh = (const float*)d_in[4];
    const float* super_Wc = (const float*)d_in[5];
    const float* super_b  = (const float*)d_in[6];
    const float* final_W  = (const float*)d_in[7];
    const float* final_b  = (const float*)d_in[8];

    float* out   = (float*)d_out;
    float* probs = out + (size_t)B_TOK * OUT_;   // tuple order: (final_output, gate_probs)

    // workspace: bf16 weights + combine weights (~9.7 MB)
    char* ws = (char*)d_ws;
    const size_t W1_BYTES = (size_t)NEXP * HALF_ * KTOT * 2;   // 4,456,448
    const size_t W2_BYTES = (size_t)NEXP * OUT_ * HALF_ * 2;   // 4,194,304
    __bf16* w1 = (__bf16*)ws;
    __bf16* w2 = (__bf16*)(ws + W1_BYTES);
    float*  cwb = (float*)(ws + W1_BYTES + W2_BYTES);          // [B][4]

    prep_weights<<<4096, 256, 0, stream>>>(super_Wh, super_Wc, final_W, w1, w2);
    gate_kernel<<<B_TOK / 8, 256, 0, stream>>>(hidden, gate_W, gate_b, probs, cwb);
    moe_main<<<B_TOK / BM, 256, SMEM_BYTES, stream>>>(hidden, context, w1, w2,
                                                      super_b, final_b, cwb, out);
}
